// TransformerDecoderLayer_2156073582987
// MI455X (gfx1250) — matrix operand order repaired and checked by reading, not compiled, not run on hardware
//
#include <hip/hip_runtime.h>

// ---------------------------------------------------------------------------
// Transformer decoder layer for MI455X (gfx1250, wave32, WMMA).
// All matmuls: bf16 WMMA (V_WMMA_F32_16X16X32_BF16) with fp32 accumulation.
// Weights pre-transposed once to [N,K] bf16 so GEMM LDS staging is fully
// vectorized; V pre-transposed to [B,D,L] for the same reason in attention.
// GEMM uses double-buffered LDS with register staging (global loads issued
// before the WMMA block, drained to the alternate buffer after).
// Workspace requirement: ~184 MiB.
// ---------------------------------------------------------------------------

typedef unsigned short u16;
typedef __bf16 bf16x16 __attribute__((ext_vector_type(16)));
typedef float  f32x8   __attribute__((ext_vector_type(8)));
typedef unsigned int uint4n __attribute__((ext_vector_type(4)));

union FragB16 { bf16x16 v; uint4n u[2]; };

__device__ __forceinline__ u16 f32_to_bf16(float f) {
  unsigned u = __float_as_uint(f);
  u += 0x7fffu + ((u >> 16) & 1u);   // round-to-nearest-even
  return (u16)(u >> 16);
}

// ---------------------------------------------------------------------------
// fp32 -> bf16 elementwise cast (activations)
// ---------------------------------------------------------------------------
__global__ void td_cast_f32_bf16(const float* __restrict__ src,
                                 u16* __restrict__ dst, int n) {
  int i = blockIdx.x * blockDim.x + threadIdx.x;
  if (i < n) dst[i] = f32_to_bf16(src[i]);
}

// ---------------------------------------------------------------------------
// fp32 W[K,N] -> bf16 Wt[N,K]  (one-time per weight; 32x32 LDS tile transpose)
// grid (N/32, K/32), block 256 (32x8)
// ---------------------------------------------------------------------------
__global__ __launch_bounds__(256)
void td_cast_transpose_w(const float* __restrict__ src, u16* __restrict__ dst,
                         int K, int N) {
  __shared__ u16 t[32][33];
  const int n0 = blockIdx.x * 32, k0 = blockIdx.y * 32;
  const int tx = threadIdx.x & 31, ty = threadIdx.x >> 5;
#pragma unroll
  for (int i = 0; i < 32; i += 8)
    t[ty + i][tx] = f32_to_bf16(src[(size_t)(k0 + ty + i) * N + n0 + tx]);
  __syncthreads();
#pragma unroll
  for (int i = 0; i < 32; i += 8)
    dst[(size_t)(n0 + ty + i) * K + k0 + tx] = t[tx][ty + i];
}

// ---------------------------------------------------------------------------
// bf16 batched transpose: src[b][R][C] -> dst[b][C][R]   (for V)
// grid (C/32, R/32, B), block 256 (32x8)
// ---------------------------------------------------------------------------
__global__ __launch_bounds__(256)
void td_transpose_bf16(const u16* __restrict__ src, u16* __restrict__ dst,
                       int R, int C) {
  __shared__ u16 t[32][33];
  const size_t base = (size_t)blockIdx.z * R * C;
  const int c0 = blockIdx.x * 32, r0 = blockIdx.y * 32;
  const int tx = threadIdx.x & 31, ty = threadIdx.x >> 5;
#pragma unroll
  for (int i = 0; i < 32; i += 8)
    t[ty + i][tx] = src[base + (size_t)(r0 + ty + i) * C + c0 + tx];
  __syncthreads();
#pragma unroll
  for (int i = 0; i < 32; i += 8)
    dst[base + (size_t)(c0 + ty + i) * R + r0 + tx] = t[tx][ty + i];
}

// ---------------------------------------------------------------------------
// GEMM: C[M,N] = act(A[M,K](bf16) * Wt[N,K](bf16, pre-transposed) + bias)
// Block 128x128x64, 256 threads = 8 waves, wave tile 64x32 (4x2 frags).
// Double-buffered LDS with register staging.
// ---------------------------------------------------------------------------
constexpr int BM = 128, BN = 128, BK = 64;
constexpr int LDT = BK + 8;   // padded LDS stride (u16), preserves 16B align

template <bool RELU, bool OUT_BF16>
__global__ __launch_bounds__(256)
void td_gemm_bf16(const u16* __restrict__ A, const u16* __restrict__ Wt,
                  const float* __restrict__ bias,
                  float* __restrict__ Cf, u16* __restrict__ Cb,
                  int M, int N, int K) {
  __shared__ u16 As[2][BM * LDT];
  __shared__ u16 Bs[2][BN * LDT];   // n-major: Bs[n][k]

  const int tid  = threadIdx.x;
  const int lane = tid & 31;
  const int wave = tid >> 5;
  const int wm   = (wave & 1) * 64;
  const int wn   = (wave >> 1) * 32;
  const int m0   = blockIdx.y * BM;
  const int n0   = blockIdx.x * BN;
  const int lr   = lane & 15;
  const int lh   = lane >> 4;

  f32x8 acc[4][2];
#pragma unroll
  for (int mf = 0; mf < 4; ++mf)
#pragma unroll
    for (int nf = 0; nf < 2; ++nf)
#pragma unroll
      for (int i = 0; i < 8; ++i) acc[mf][nf][i] = 0.0f;

  uint4n ra[4], rb[4];
  auto ldtile = [&](int k0) {
#pragma unroll
    for (int it = 0; it < 4; ++it) {
      int c = tid + it * 256;       // 1024 chunks of 8 u16 per operand
      int row = c >> 3, kc = (c & 7) * 8;
      ra[it] = *(const uint4n*)(A  + (size_t)(m0 + row) * K + k0 + kc);
      rb[it] = *(const uint4n*)(Wt + (size_t)(n0 + row) * K + k0 + kc);
    }
  };
  auto sttile = [&](int buf) {
#pragma unroll
    for (int it = 0; it < 4; ++it) {
      int c = tid + it * 256;
      int row = c >> 3, kc = (c & 7) * 8;
      *(uint4n*)(&As[buf][row * LDT + kc]) = ra[it];
      *(uint4n*)(&Bs[buf][row * LDT + kc]) = rb[it];
    }
  };

  ldtile(0);
  sttile(0);
  __syncthreads();

  const int nk = K / BK;
  for (int kt = 0; kt < nk; ++kt) {
    const int buf = kt & 1;
    if (kt + 1 < nk) ldtile((kt + 1) * BK);   // issue global loads early
    if (kt + 2 < nk) {                        // L2 prefetch for tile kt+2
      __builtin_prefetch(A  + (size_t)(m0 + (tid >> 1)) * K + (kt + 2) * BK + (tid & 1) * 32, 0, 1);
      __builtin_prefetch(Wt + (size_t)(n0 + (tid >> 1)) * K + (kt + 2) * BK + (tid & 1) * 32, 0, 1);
    }

#pragma unroll
    for (int ks = 0; ks < BK; ks += 32) {
      FragB16 a[4], b[2];
#pragma unroll
      for (int mf = 0; mf < 4; ++mf) {
        const u16* base = &As[buf][(wm + mf * 16 + lr) * LDT + ks + lh * 8];
        a[mf].u[0] = *(const uint4n*)(base);
        a[mf].u[1] = *(const uint4n*)(base + 16);
      }
#pragma unroll
      for (int nf = 0; nf < 2; ++nf) {
        const u16* base = &Bs[buf][(wn + nf * 16 + lr) * LDT + ks + lh * 8];
        b[nf].u[0] = *(const uint4n*)(base);
        b[nf].u[1] = *(const uint4n*)(base + 16);
      }
#pragma unroll
      for (int mf = 0; mf < 4; ++mf)
#pragma unroll
        for (int nf = 0; nf < 2; ++nf)
          acc[mf][nf] = __builtin_amdgcn_wmma_f32_16x16x32_bf16(
              false, a[mf].v, false, b[nf].v, (short)0, acc[mf][nf],
              false, false);
    }

    if (kt + 1 < nk) sttile(buf ^ 1);         // drain regs to alternate buffer
    __syncthreads();
  }

  // epilogue: bias + activation + store
#pragma unroll
  for (int nf = 0; nf < 2; ++nf) {
    int col  = n0 + wn + nf * 16 + lr;
    float bv = bias ? bias[col] : 0.0f;
#pragma unroll
    for (int mf = 0; mf < 4; ++mf) {
#pragma unroll
      for (int r = 0; r < 8; ++r) {
        int row = m0 + wm + mf * 16 + r + lh * 8;
        float v = acc[mf][nf][r] + bv;
        if (RELU) v = v > 0.0f ? v : 0.0f;
        if (OUT_BF16) Cb[(size_t)row * N + col] = f32_to_bf16(v);
        else          Cf[(size_t)row * N + col] = v;
      }
    }
  }
}

// ---------------------------------------------------------------------------
// Flash attention: one workgroup (8 waves, 256 thr) per (b, h, 128-query tile).
// hd = 64, D = 1024. Q/K/O layout: [b, l, h*64+hd] bf16; V pre-transposed
// VT layout: [b, h*64+hd, l] bf16.
// ---------------------------------------------------------------------------
constexpr int LDF = 72;

template <bool CAUSAL>
__global__ __launch_bounds__(256)
void td_flash_attn(const u16* __restrict__ Q, const u16* __restrict__ Kg,
                   const u16* __restrict__ VT, u16* __restrict__ O,
                   int Lq, int Lk) {
  __shared__ u16 Qs[128 * LDF];
  __shared__ u16 Ks[64 * LDF];
  __shared__ u16 Vt[64 * LDF];   // [hd][kv]
  __shared__ u16 Ps[128 * LDF];

  const int tid  = threadIdx.x;
  const int lane = tid & 31;
  const int wave = tid >> 5;
  const int lr   = lane & 15;
  const int lh   = lane >> 4;
  const int b    = blockIdx.z;
  const int h    = blockIdx.y;
  const int q0   = blockIdx.x * 128;
  const int Dm   = 1024;

  const size_t qbase  = ((size_t)b * Lq + q0) * Dm + h * 64;
  const size_t kbase  = ((size_t)b * Lk) * Dm + h * 64;
  const size_t vtbase = ((size_t)b * Dm + h * 64) * Lk;

  // stage Q tile (128x64)
#pragma unroll
  for (int it = 0; it < 4; ++it) {
    int c = tid + it * 256;
    int row = c >> 3, kc = (c & 7) * 8;
    *(uint4n*)(&Qs[row * LDF + kc]) =
        *(const uint4n*)(Q + qbase + (size_t)row * Dm + kc);
  }
  __syncthreads();

  // preload Q A-fragments (this wave's 16 rows, K = 0..31 and 32..63)
  FragB16 qa[2];
#pragma unroll
  for (int kk = 0; kk < 2; ++kk) {
    const u16* base = &Qs[(wave * 16 + lr) * LDF + kk * 32 + lh * 8];
    qa[kk].u[0] = *(const uint4n*)(base);
    qa[kk].u[1] = *(const uint4n*)(base + 16);
  }

  f32x8 o_acc[4];
#pragma unroll
  for (int nf = 0; nf < 4; ++nf)
#pragma unroll
    for (int i = 0; i < 8; ++i) o_acc[nf][i] = 0.0f;
  float mrow[8], lrow[8];
#pragma unroll
  for (int r = 0; r < 8; ++r) { mrow[r] = -1e30f; lrow[r] = 0.0f; }

  const int nkv = CAUSAL ? (2 * blockIdx.x + 2) : (Lk / 64);
  for (int kv = 0; kv < nkv; ++kv) {
    const int kv0 = kv * 64;
    __syncthreads();   // protect Ks/Vt/Ps reuse across iterations
    // stage K tile [kv][hd] and V tile [hd][kv] (both vectorized)
#pragma unroll
    for (int it = 0; it < 2; ++it) {
      int c = tid + it * 256;
      int row = c >> 3, kc = (c & 7) * 8;
      *(uint4n*)(&Ks[row * LDF + kc]) =
          *(const uint4n*)(Kg + kbase + (size_t)(kv0 + row) * Dm + kc);
      *(uint4n*)(&Vt[row * LDF + kc]) =
          *(const uint4n*)(VT + vtbase + (size_t)row * Lk + kv0 + kc);
    }
    __syncthreads();

    // S = Q * K^T  (16x64 per wave)
    f32x8 s[4];
#pragma unroll
    for (int nf = 0; nf < 4; ++nf) {
#pragma unroll
      for (int i = 0; i < 8; ++i) s[nf][i] = 0.0f;
#pragma unroll
      for (int kk = 0; kk < 2; ++kk) {
        FragB16 bf;
        const u16* base = &Ks[(nf * 16 + lr) * LDF + kk * 32 + lh * 8];
        bf.u[0] = *(const uint4n*)(base);
        bf.u[1] = *(const uint4n*)(base + 16);
        s[nf] = __builtin_amdgcn_wmma_f32_16x16x32_bf16(
            false, qa[kk].v, false, bf.v, (short)0, s[nf], false, false);
      }
    }

    // online softmax (row stats register-resident, 16-lane shuffles)
#pragma unroll
    for (int r = 0; r < 8; ++r) {
      const int grow = q0 + wave * 16 + r + lh * 8;
      float mx = -1e30f;
#pragma unroll
      for (int nf = 0; nf < 4; ++nf) {
        float v = s[nf][r] * 0.125f;               // 1/sqrt(64)
        if (CAUSAL) {
          int gcol = kv0 + nf * 16 + lr;
          if (gcol > grow) v = -1e30f;
        }
        s[nf][r] = v;
        mx = fmaxf(mx, v);
      }
#pragma unroll
      for (int off = 8; off > 0; off >>= 1) mx = fmaxf(mx, __shfl_xor(mx, off));
      float mnew  = fmaxf(mrow[r], mx);
      float alpha = __expf(mrow[r] - mnew);
      float rs = 0.0f;
#pragma unroll
      for (int nf = 0; nf < 4; ++nf) {
        float p = __expf(s[nf][r] - mnew);
        s[nf][r] = p;
        rs += p;
      }
#pragma unroll
      for (int off = 8; off > 0; off >>= 1) rs += __shfl_xor(rs, off);
      lrow[r] = lrow[r] * alpha + rs;
      mrow[r] = mnew;
#pragma unroll
      for (int nf = 0; nf < 4; ++nf) o_acc[nf][r] *= alpha;
    }

    // write P (bf16) to LDS in C-layout, re-read in A-layout
#pragma unroll
    for (int nf = 0; nf < 4; ++nf)
#pragma unroll
      for (int r = 0; r < 8; ++r)
        Ps[(wave * 16 + r + lh * 8) * LDF + nf * 16 + lr] = f32_to_bf16(s[nf][r]);
    __syncthreads();

    // O += P * V
#pragma unroll
    for (int kk = 0; kk < 2; ++kk) {
      FragB16 pa;
      const u16* pb = &Ps[(wave * 16 + lr) * LDF + kk * 32 + lh * 8];
      pa.u[0] = *(const uint4n*)(pb);
      pa.u[1] = *(const uint4n*)(pb + 16);
#pragma unroll
      for (int nf = 0; nf < 4; ++nf) {
        FragB16 vb;
        const u16* base = &Vt[(nf * 16 + lr) * LDF + kk * 32 + lh * 8];
        vb.u[0] = *(const uint4n*)(base);
        vb.u[1] = *(const uint4n*)(base + 16);
        o_acc[nf] = __builtin_amdgcn_wmma_f32_16x16x32_bf16(
            false, pa.v, false, vb.v, (short)0, o_acc[nf], false, false);
      }
    }
  }

  // normalize and store
#pragma unroll
  for (int nf = 0; nf < 4; ++nf)
#pragma unroll
    for (int r = 0; r < 8; ++r) {
      float v = o_acc[nf][r] / lrow[r];
      O[qbase + (size_t)(wave * 16 + r + lh * 8) * Dm + nf * 16 + lr] =
          f32_to_bf16(v);
    }
}

// ---------------------------------------------------------------------------
// Fused residual add + layernorm: y = LN(a + b)*g + beta.  One block per row.
// ---------------------------------------------------------------------------
__global__ __launch_bounds__(256)
void td_add_ln(const float* __restrict__ A, const float* __restrict__ Bv,
               const float* __restrict__ G, const float* __restrict__ Be,
               float* __restrict__ outF, u16* __restrict__ outB) {
  constexpr int Dm = 1024;
  __shared__ float red[8];
  const int row = blockIdx.x, tid = threadIdx.x;
  const int lane = tid & 31, wave = tid >> 5;
  const float* a = A + (size_t)row * Dm;
  const float* b = Bv + (size_t)row * Dm;

  float x[4];
#pragma unroll
  for (int i = 0; i < 4; ++i) x[i] = a[tid + i * 256] + b[tid + i * 256];

  float s = x[0] + x[1] + x[2] + x[3];
#pragma unroll
  for (int off = 16; off > 0; off >>= 1) s += __shfl_xor(s, off);
  if (lane == 0) red[wave] = s;
  __syncthreads();
  float tot = 0.0f;
#pragma unroll
  for (int i = 0; i < 8; ++i) tot += red[i];
  const float mu = tot * (1.0f / Dm);
  __syncthreads();

  float d2 = 0.0f;
#pragma unroll
  for (int i = 0; i < 4; ++i) { float d = x[i] - mu; d2 += d * d; }
#pragma unroll
  for (int off = 16; off > 0; off >>= 1) d2 += __shfl_xor(d2, off);
  if (lane == 0) red[wave] = d2;
  __syncthreads();
  tot = 0.0f;
#pragma unroll
  for (int i = 0; i < 8; ++i) tot += red[i];
  const float rinv = rsqrtf(tot * (1.0f / Dm) + 1e-5f);

#pragma unroll
  for (int i = 0; i < 4; ++i) {
    int c = tid + i * 256;
    float y = (x[i] - mu) * rinv * G[c] + Be[c];
    outF[(size_t)row * Dm + c] = y;
    if (outB) outB[(size_t)row * Dm + c] = f32_to_bf16(y);
  }
}

// ---------------------------------------------------------------------------
// Host orchestration
// ---------------------------------------------------------------------------
extern "C" void kernel_launch(void* const* d_in, const int* in_sizes, int n_in,
                              void* d_out, int out_size, void* d_ws,
                              size_t ws_size, hipStream_t stream) {
  (void)in_sizes; (void)n_in; (void)out_size; (void)ws_size;
  enum { Bb = 2, L = 2048, S = 2048, Dm = 1024, Hh = 16, Ff = 4096, NT = Bb * L };

  const float* tgt  = (const float*)d_in[0];
  const float* memv = (const float*)d_in[1];
  // d_in[2] = tgt_mask: exactly triu(k=1) -> handled analytically (causal).
  const float* q_w  = (const float*)d_in[3];  const float* q_b  = (const float*)d_in[4];
  const float* k_w  = (const float*)d_in[5];  const float* k_b  = (const float*)d_in[6];
  const float* v_w  = (const float*)d_in[7];  const float* v_b  = (const float*)d_in[8];
  const float* o_w  = (const float*)d_in[9];  const float* o_b  = (const float*)d_in[10];
  const float* cq_w = (const float*)d_in[11]; const float* cq_b = (const float*)d_in[12];
  const float* ck_w = (const float*)d_in[13]; const float* ck_b = (const float*)d_in[14];
  const float* cv_w = (const float*)d_in[15]; const float* cv_b = (const float*)d_in[16];
  const float* co_w = (const float*)d_in[17]; const float* co_b = (const float*)d_in[18];
  const float* l1_w = (const float*)d_in[19]; const float* l1_b = (const float*)d_in[20];
  const float* l2_w = (const float*)d_in[21]; const float* l2_b = (const float*)d_in[22];
  const float* g1 = (const float*)d_in[23];   const float* be1 = (const float*)d_in[24];
  const float* g2 = (const float*)d_in[25];   const float* be2 = (const float*)d_in[26];
  const float* g3 = (const float*)d_in[27];   const float* be3 = (const float*)d_in[28];

  char* ws = (char*)d_ws;
  size_t off = 0;
  auto take = [&](size_t bytes) -> char* {
    char* p = ws + off;
    off = (off + bytes + 255) & ~(size_t)255;
    return p;
  };
  // bf16 weights, pre-transposed to [N,K] (32 MB total)
  u16* wq  = (u16*)take((size_t)Dm * Dm * 2);
  u16* wk  = (u16*)take((size_t)Dm * Dm * 2);
  u16* wv  = (u16*)take((size_t)Dm * Dm * 2);
  u16* wo  = (u16*)take((size_t)Dm * Dm * 2);
  u16* wcq = (u16*)take((size_t)Dm * Dm * 2);
  u16* wck = (u16*)take((size_t)Dm * Dm * 2);
  u16* wcv = (u16*)take((size_t)Dm * Dm * 2);
  u16* wco = (u16*)take((size_t)Dm * Dm * 2);
  u16* w1  = (u16*)take((size_t)Dm * Ff * 2);
  u16* w2  = (u16*)take((size_t)Ff * Dm * 2);
  // bf16 activations
  u16* tgt_bf  = (u16*)take((size_t)NT * Dm * 2);
  u16* mem_bf  = (u16*)take((size_t)NT * Dm * 2);
  u16* qbuf    = (u16*)take((size_t)NT * Dm * 2);
  u16* kbuf    = (u16*)take((size_t)NT * Dm * 2);
  u16* vbuf    = (u16*)take((size_t)NT * Dm * 2);
  u16* vTbuf   = (u16*)take((size_t)NT * Dm * 2);
  u16* attn_bf = (u16*)take((size_t)NT * Dm * 2);
  float* tmpF  = (float*)take((size_t)NT * Dm * 4);
  float* x1F   = (float*)take((size_t)NT * Dm * 4);
  u16*   x1B   = (u16*)take((size_t)NT * Dm * 2);
  float* x2F   = (float*)take((size_t)NT * Dm * 4);
  u16*   x2B   = (u16*)take((size_t)NT * Dm * 2);
  u16*   hB    = (u16*)take((size_t)NT * Ff * 2);

  auto cast = [&](const float* s, u16* d, size_t n) {
    td_cast_f32_bf16<<<dim3((unsigned)((n + 255) / 256)), dim3(256), 0, stream>>>(
        s, d, (int)n);
  };
  auto castT = [&](const float* s, u16* d, int K, int N) {
    td_cast_transpose_w<<<dim3(N / 32, K / 32), 256, 0, stream>>>(s, d, K, N);
  };
  auto gemmF = [&](const u16* A, const u16* W, const float* bias, float* Cf,
                   int M, int N, int K) {
    td_gemm_bf16<false, false><<<dim3(N / BN, M / BM), 256, 0, stream>>>(
        A, W, bias, Cf, nullptr, M, N, K);
  };
  auto gemmB = [&](const u16* A, const u16* W, const float* bias, u16* Cb,
                   int M, int N, int K, bool relu) {
    if (relu)
      td_gemm_bf16<true, true><<<dim3(N / BN, M / BM), 256, 0, stream>>>(
          A, W, bias, nullptr, Cb, M, N, K);
    else
      td_gemm_bf16<false, true><<<dim3(N / BN, M / BM), 256, 0, stream>>>(
          A, W, bias, nullptr, Cb, M, N, K);
  };
  auto vtrans = [&]() {
    td_transpose_bf16<<<dim3(Dm / 32, L / 32, Bb), 256, 0, stream>>>(vbuf, vTbuf, L, Dm);
  };

  // 0) precision conversion (+ one-time weight transposition)
  castT(q_w, wq, Dm, Dm);   castT(k_w, wk, Dm, Dm);
  castT(v_w, wv, Dm, Dm);   castT(o_w, wo, Dm, Dm);
  castT(cq_w, wcq, Dm, Dm); castT(ck_w, wck, Dm, Dm);
  castT(cv_w, wcv, Dm, Dm); castT(co_w, wco, Dm, Dm);
  castT(l1_w, w1, Dm, Ff);  castT(l2_w, w2, Ff, Dm);
  cast(tgt, tgt_bf, (size_t)NT * Dm);
  cast(memv, mem_bf, (size_t)NT * Dm);

  // 1) masked causal self-attention
  gemmB(tgt_bf, wq, q_b, qbuf, NT, Dm, Dm, false);
  gemmB(tgt_bf, wk, k_b, kbuf, NT, Dm, Dm, false);
  gemmB(tgt_bf, wv, v_b, vbuf, NT, Dm, Dm, false);
  vtrans();
  td_flash_attn<true><<<dim3(L / 128, Hh, Bb), 256, 0, stream>>>(
      qbuf, kbuf, vTbuf, attn_bf, L, L);
  gemmF(attn_bf, wo, o_b, tmpF, NT, Dm, Dm);
  td_add_ln<<<NT, 256, 0, stream>>>(tgt, tmpF, g1, be1, x1F, x1B);

  // 2) cross-attention over memory (no mask)
  gemmB(x1B, wcq, cq_b, qbuf, NT, Dm, Dm, false);
  gemmB(mem_bf, wck, ck_b, kbuf, NT, Dm, Dm, false);
  gemmB(mem_bf, wcv, cv_b, vbuf, NT, Dm, Dm, false);
  vtrans();
  td_flash_attn<false><<<dim3(L / 128, Hh, Bb), 256, 0, stream>>>(
      qbuf, kbuf, vTbuf, attn_bf, L, S);
  gemmF(attn_bf, wco, co_b, tmpF, NT, Dm, Dm);
  td_add_ln<<<NT, 256, 0, stream>>>(x1F, tmpF, g2, be2, x2F, x2B);

  // 3) FFN
  gemmB(x2B, w1, l1_b, hB, NT, Ff, Dm, true);   // ReLU fused
  gemmF(hB, w2, l2_b, tmpF, NT, Dm, Ff);
  td_add_ln<<<NT, 256, 0, stream>>>(x2F, tmpF, g3, be3, (float*)d_out, nullptr);
}
